// VoteNet_55130200211569
// MI455X (gfx1250) — matrix operand order repaired and checked by reading, not compiled, not run on hardware
//
#include <hip/hip_runtime.h>
#include <hip/hip_bf16.h>
#include <stdint.h>

typedef __attribute__((ext_vector_type(16))) _Float16 v16h;
typedef __attribute__((ext_vector_type(8)))  _Float16 v8h;
typedef __attribute__((ext_vector_type(8)))  float    v8f;

// ---------------------------------------------------------------------------
// CDNA5 async global->LDS path. The builtins exist on this toolchain (guard
// passed in round 2); their params are typed address-space pointers:
//   b128: (as1 v4i* src, as3 v4i* dst, imm offset, imm cpol)
//   b32 : (as1 int* src, as3 int* dst, imm offset, imm cpol)
// We build exact pointer types via integer casts (generic shared address keeps
// the LDS offset in its low 32 bits; global generic is identity-mapped).
// ---------------------------------------------------------------------------
#if defined(__gfx1250__) &&                                            \
    __has_builtin(__builtin_amdgcn_global_load_async_to_lds_b128) &&   \
    __has_builtin(__builtin_amdgcn_global_load_async_to_lds_b32)  &&   \
    __has_builtin(__builtin_amdgcn_s_wait_asynccnt)
#define USE_ASYNC_LDS 1
#else
#define USE_ASYNC_LDS 0
#endif

#if USE_ASYNC_LDS
typedef int vs4i __attribute__((vector_size(16)));
typedef __attribute__((address_space(1))) vs4i as1_vs4i;
typedef __attribute__((address_space(3))) vs4i as3_vs4i;
typedef __attribute__((address_space(1))) int  as1_int;
typedef __attribute__((address_space(3))) int  as3_int;

static __device__ __forceinline__ void async_lds_b128(const void* g, const void* l) {
  __builtin_amdgcn_global_load_async_to_lds_b128(
      (as1_vs4i*)(uintptr_t)g, (as3_vs4i*)(uint32_t)(uintptr_t)l, 0, 0);
}
static __device__ __forceinline__ void async_lds_b32(const void* g, const void* l) {
  __builtin_amdgcn_global_load_async_to_lds_b32(
      (as1_int*)(uintptr_t)g, (as3_int*)(uint32_t)(uintptr_t)l, 0, 0);
}
#endif

// ---------------------------------------------------------------------------
// Weight conversion: W f32 [K,N] row-major -> Wt f16 [Npad, Kpad] (transposed,
// zero padded) so GEMM B-fragment loads are contiguous.
// ---------------------------------------------------------------------------
__global__ void convert_wt_kernel(const float* __restrict__ W, _Float16* __restrict__ Wt,
                                  int K, int N, int Kpad, int Npad) {
  int total = Kpad * Npad;
  for (int g = blockIdx.x * blockDim.x + threadIdx.x; g < total; g += gridDim.x * blockDim.x) {
    int k = g % Kpad, n = g / Kpad;
    float v = (k < K && n < N) ? W[k * N + n] : 0.f;
    Wt[g] = (_Float16)v;
  }
}

// ---------------------------------------------------------------------------
// WMMA GEMM: Out[M,Npad] = act(A[M,Kpad] @ W + bias)
//   A  : f16 row-major, Kpad % 32 == 0, M % 128 == 0
//   Wt : f16 [Npad, Kpad] (transposed weights), Npad % (16*NTILES) == 0
// Block = 256 threads (8 wave32 waves). Wave w computes rows
// [blockIdx.x*128 + w*16, +16) x cols [blockIdx.y*16*NTILES, +16*NTILES):
// the A fragment is reused for NTILES WMMAs. The W tile (16*NTILES cols x 32 k)
// is double-buffered in LDS: tile k+1 is staged via async global->LDS DMA
// (ASYNCcnt) while tile k feeds the WMMAs -> one barrier per k-step.
// ---------------------------------------------------------------------------
template <bool RELU, int NTILES>
__global__ __launch_bounds__(256) void gemm_wmma_kernel(
    const _Float16* __restrict__ A, const _Float16* __restrict__ Wt,
    const float* __restrict__ bias, _Float16* __restrict__ Out,
    int Kpad, int Npad, int Nreal) {
  constexpr int TILE_HALF = NTILES * 16 * 32;  // f16 per LDS buffer
  __shared__ __align__(128) _Float16 ldsW[2 * TILE_HALF];
  const int tid = threadIdx.x;
  const int wave = tid >> 5;
  const int lane = tid & 31;
  const int half = lane >> 4;  // 16-lane half of the wave
  const int ln16 = lane & 15;
  const int tileN = blockIdx.y * (16 * NTILES);
  const int m = blockIdx.x * 128 + wave * 16 + ln16;
  const int nK = Kpad >> 5;

  // per-thread staging coordinates (NTILES==4: 16B each, NTILES==1: 4B each)
  int sr, sc;
  if (NTILES == 4) { sr = tid >> 2; sc = (tid & 3) * 8; }
  else             { sr = tid >> 4; sc = (tid & 15) * 2; }

  // ---- prologue: stage k-tile 0 into buffer 0 ----
  {
    const _Float16* src = Wt + (size_t)(tileN + sr) * Kpad + sc;
    _Float16* dst = &ldsW[sr * 32 + sc];
#if USE_ASYNC_LDS
    if (NTILES == 4) async_lds_b128(src, dst);
    else             async_lds_b32(src, dst);
    __builtin_amdgcn_s_wait_asynccnt(0);
#else
    if (NTILES == 4) *(uint4*)dst = *(const uint4*)src;
    else             *(uint32_t*)dst = *(const uint32_t*)src;
#endif
  }
  __syncthreads();

  v8f acc[NTILES] = {};
  for (int kt = 0; kt < nK; ++kt) {
    const int k0 = kt << 5;
    const int kn = (kt + 1 < nK) ? (k0 + 32) : k0;  // clamped (branchless tail)
    const int cbuf = kt & 1, nbuf = (kt + 1) & 1;

    // stage next W tile into the other buffer (overlaps with compute below)
    const _Float16* src = Wt + (size_t)(tileN + sr) * Kpad + kn + sc;
    _Float16* dst = &ldsW[nbuf * TILE_HALF + sr * 32 + sc];
#if USE_ASYNC_LDS
    if (NTILES == 4) async_lds_b128(src, dst);
    else             async_lds_b32(src, dst);
#else
    uint4 wreg;
    if (NTILES == 4) wreg = *(const uint4*)src;
    else             wreg.x = *(const uint32_t*)src;
#endif

    // A fragment (16-bit A 16x32 layout): half=0 -> K 0..7 | 16..23,
    // half=1 -> K 8..15 | 24..31 : two contiguous 16B loads, reused NTILES x.
    const _Float16* arow = A + (size_t)m * Kpad + k0;
    __builtin_prefetch(arow + 64, 0, 1);
    v8h alo = *(const v8h*)(arow + 8 * half);
    v8h ahi = *(const v8h*)(arow + 16 + 8 * half);
    v16h afrag;
#pragma unroll
    for (int t = 0; t < 8; ++t) { afrag[t] = alo[t]; afrag[8 + t] = ahi[t]; }

    const _Float16* wb = &ldsW[cbuf * TILE_HALF];
#pragma unroll
    for (int nt = 0; nt < NTILES; ++nt) {
      // B fragment: lane holds column; elements e=0..7 <-> K = 8*half + e, e=8..15 <-> K = 16 + 8*half + (e-8).
      v8h blo = *(const v8h*)&wb[(nt * 16 + ln16) * 32 + half * 8];
      v8h bhi = *(const v8h*)&wb[(nt * 16 + ln16) * 32 + 16 + half * 8];
      v16h bfrag;
#pragma unroll
      for (int t = 0; t < 8; ++t) { bfrag[t] = blo[t]; bfrag[8 + t] = bhi[t]; }
      acc[nt] = __builtin_amdgcn_wmma_f32_16x16x32_f16(
          false, afrag, false, bfrag, (short)0, acc[nt], false, false);
    }

#if USE_ASYNC_LDS
    __builtin_amdgcn_s_wait_asynccnt(0);
#else
    if (NTILES == 4) *(uint4*)dst = wreg;
    else             *(uint32_t*)dst = wreg.x;
#endif
    __syncthreads();
  }

#pragma unroll
  for (int nt = 0; nt < NTILES; ++nt) {
    int n = tileN + nt * 16 + ln16;
    float bn = (n < Nreal) ? bias[n] : 0.f;
#pragma unroll
    for (int r = 0; r < 8; ++r) {
      float v = acc[nt][r] + bn;
      if (RELU) v = fmaxf(v, 0.f);
      int row = blockIdx.x * 128 + wave * 16 + half * 8 + r;  // C layout: M = r + 8*half
      Out[(size_t)row * Npad + n] = (_Float16)v;
    }
  }
}

// ---------------------------------------------------------------------------
// Farthest point sampling: one block (one WGP, 32 wave32 waves) per batch,
// npoint serial iterations with a block-wide argmax tree reduction.
// ---------------------------------------------------------------------------
__global__ __launch_bounds__(1024) void fps_kernel(const float* __restrict__ xyz, int N,
                                                   int npoint, int* __restrict__ outIdx,
                                                   float* __restrict__ dists) {
  const int b = blockIdx.x;
  const int tid = threadIdx.x;
  const float* px = xyz + (size_t)b * N * 3;
  float* pd = dists + (size_t)b * N;
  for (int i = tid; i < N; i += 1024) pd[i] = 1e10f;
  __shared__ float sv[1024];
  __shared__ int si[1024];
  __syncthreads();
  int far = 0;
  for (int s = 0; s < npoint; ++s) {
    if (tid == 0) outIdx[b * npoint + s] = far;
    float cx = px[far * 3 + 0], cy = px[far * 3 + 1], cz = px[far * 3 + 2];
    float best = -1.f;
    int bi = 0;
    for (int i = tid; i < N; i += 1024) {
      float dx = px[i * 3 + 0] - cx, dy = px[i * 3 + 1] - cy, dz = px[i * 3 + 2] - cz;
      float d = dx * dx + dy * dy + dz * dz;
      float nd = fminf(pd[i], d);
      pd[i] = nd;
      if (nd > best) { best = nd; bi = i; }
    }
    sv[tid] = best;
    si[tid] = bi;
    __syncthreads();
    for (int o = 512; o > 0; o >>= 1) {
      if (tid < o && sv[tid + o] > sv[tid]) { sv[tid] = sv[tid + o]; si[tid] = si[tid + o]; }
      __syncthreads();
    }
    far = si[0];
    __syncthreads();
  }
}

__global__ void gather_xyz_kernel(const float* __restrict__ xyz, const int* __restrict__ idx,
                                  float* __restrict__ out, int NP, int Nsrc, int total) {
  for (int g = blockIdx.x * blockDim.x + threadIdx.x; g < total; g += gridDim.x * blockDim.x) {
    int k = g % 3, rem = g / 3;
    int i = rem % NP, b = rem / NP;
    out[g] = xyz[((size_t)b * Nsrc + idx[b * NP + i]) * 3 + k];
  }
}

// Ball query: reference sorts (idx if d2<=r2 else N) ascending and takes the
// first nsample -> identical to scanning in index order taking first hits.
__global__ void ball_query_kernel(const float* __restrict__ xyz, const float* __restrict__ nxyz,
                                  int Nsrc, int NP, int NS, float r2,
                                  int* __restrict__ nidx, int totalQ) {
  int q = blockIdx.x * blockDim.x + threadIdx.x;
  if (q >= totalQ) return;
  int b = q / NP;
  float nx = nxyz[q * 3 + 0], ny = nxyz[q * 3 + 1], nz = nxyz[q * 3 + 2];
  const float* base = xyz + (size_t)b * Nsrc * 3;
  int found = 0, first = Nsrc - 1;
  for (int p = 0; p < Nsrc && found < NS; ++p) {
    float dx = base[p * 3 + 0] - nx, dy = base[p * 3 + 1] - ny, dz = base[p * 3 + 2] - nz;
    if (dx * dx + dy * dy + dz * dz <= r2) {
      if (found == 0) first = p;
      nidx[(size_t)q * NS + found] = p;
      ++found;
    }
  }
  if (found == 0)
    for (int j = 0; j < NS; ++j) nidx[(size_t)q * NS + j] = Nsrc - 1;
  else
    for (int j = found; j < NS; ++j) nidx[(size_t)q * NS + j] = first;
}

// Build MLP input X f16 [B*NP*NS, Kpad]: cols 0..2 = grouped relative xyz,
// 3..3+C = gathered features, rest zero padding.
__global__ void group_kernel(const float* __restrict__ xyz, const float* __restrict__ nxyz,
                             const int* __restrict__ nidx, const _Float16* __restrict__ feat,
                             int C, int Nsrc, int NP, int NS, int Kpad,
                             _Float16* __restrict__ X, int total) {
  for (int g = blockIdx.x * blockDim.x + threadIdx.x; g < total; g += gridDim.x * blockDim.x) {
    int k = g % Kpad;
    int row = g / Kpad;
    int j = row % NS;
    int rem = row / NS;
    int i = rem % NP, b = rem / NP;
    int src = nidx[((size_t)b * NP + i) * NS + j];
    float v = 0.f;
    if (k < 3)
      v = xyz[((size_t)b * Nsrc + src) * 3 + k] - nxyz[((size_t)b * NP + i) * 3 + k];
    else if (feat && k < 3 + C)
      v = (float)feat[((size_t)b * Nsrc + src) * C + (k - 3)];
    X[g] = (_Float16)v;
  }
}

__global__ void maxpool_kernel(const _Float16* __restrict__ Y, _Float16* __restrict__ feat,
                               int NP, int NS, int C, int total) {
  for (int g = blockIdx.x * blockDim.x + threadIdx.x; g < total; g += gridDim.x * blockDim.x) {
    int c = g % C;
    int rem = g / C;
    int i = rem % NP, b = rem / NP;
    size_t base = ((size_t)(b * NP + i) * NS) * C + c;
    float m = -1e30f;
    for (int j = 0; j < NS; ++j) m = fmaxf(m, (float)Y[base + (size_t)j * C]);
    feat[((size_t)b * NP + i) * C + c] = (_Float16)m;
  }
}

// p_feat[b,t,c] = vote_feat[b, perm[t], c]  (vote rows stride 272, cols 0..255)
__global__ void permute_kernel(const _Float16* __restrict__ vote, const int* __restrict__ perm,
                               _Float16* __restrict__ pfeat, int total) {
  for (int g = blockIdx.x * blockDim.x + threadIdx.x; g < total; g += gridDim.x * blockDim.x) {
    int c = g % 256;
    int rem = g / 256;
    int t = rem % 256, b = rem / 256;
    int s = perm[t];
    pfeat[g] = vote[((size_t)(b * 512 + s)) * 272 + c];
  }
}

// Final tiny heads: obj (1), center (3), size (3) from x f16 [512,128].
__global__ void heads_kernel(const _Float16* __restrict__ x,
                             const float* __restrict__ ow, const float* __restrict__ ob,
                             const float* __restrict__ cw, const float* __restrict__ cb,
                             const float* __restrict__ sw, const float* __restrict__ sb,
                             float* __restrict__ out, int total) {
  int g = blockIdx.x * blockDim.x + threadIdx.x;
  if (g >= total) return;
  int u = g % 7, row = g / 7;
  const _Float16* xr = x + (size_t)row * 128;
  float acc = 0.f;
  if (u == 0) {
    for (int c = 0; c < 128; ++c) acc += (float)xr[c] * ow[c];
    out[row] = acc + ob[0];
  } else if (u < 4) {
    int k = u - 1;
    for (int c = 0; c < 128; ++c) acc += (float)xr[c] * cw[c * 3 + k];
    out[512 + row * 3 + k] = acc + cb[k];
  } else {
    int k = u - 4;
    for (int c = 0; c < 128; ++c) acc += (float)xr[c] * sw[c * 3 + k];
    out[512 + 1536 + row * 3 + k] = acc + sb[k];
  }
}

// ---------------------------------------------------------------------------
extern "C" void kernel_launch(void* const* d_in, const int* in_sizes, int n_in,
                              void* d_out, int out_size, void* d_ws, size_t ws_size,
                              hipStream_t stream) {
  (void)in_sizes; (void)n_in; (void)out_size; (void)ws_size;
  const float* pc = (const float*)d_in[0];
  const int* perm = (const int*)d_in[1];
  const float* sa1W[3] = {(const float*)d_in[2], (const float*)d_in[4], (const float*)d_in[6]};
  const float* sa1B[3] = {(const float*)d_in[3], (const float*)d_in[5], (const float*)d_in[7]};
  const float* sa2W[3] = {(const float*)d_in[8], (const float*)d_in[10], (const float*)d_in[12]};
  const float* sa2B[3] = {(const float*)d_in[9], (const float*)d_in[11], (const float*)d_in[13]};
  const float* sa3W[3] = {(const float*)d_in[14], (const float*)d_in[16], (const float*)d_in[18]};
  const float* sa3B[3] = {(const float*)d_in[15], (const float*)d_in[17], (const float*)d_in[19]};
  const float* voteW = (const float*)d_in[20];
  const float* voteB = (const float*)d_in[21];
  const float* hW[2] = {(const float*)d_in[22], (const float*)d_in[24]};
  const float* hB[2] = {(const float*)d_in[23], (const float*)d_in[25]};
  const float* objW = (const float*)d_in[26]; const float* objB = (const float*)d_in[27];
  const float* cenW = (const float*)d_in[28]; const float* cenB = (const float*)d_in[29];
  const float* sizW = (const float*)d_in[30]; const float* sizB = (const float*)d_in[31];

  constexpr int B = 2, N0 = 20000;

  // ---- workspace bump allocator (deterministic layout every call) ----
  size_t off = 0;
  auto alloc = [&](size_t bytes) -> char* {
    char* p = (char*)d_ws + off;
    off = (off + bytes + 255) & ~(size_t)255;
    return p;
  };
  _Float16* wt1[3] = {(_Float16*)alloc(64 * 32 * 2), (_Float16*)alloc(64 * 64 * 2),
                      (_Float16*)alloc(128 * 64 * 2)};
  _Float16* wt2[3] = {(_Float16*)alloc(128 * 160 * 2), (_Float16*)alloc(128 * 128 * 2),
                      (_Float16*)alloc(256 * 128 * 2)};
  _Float16* wt3[3] = {(_Float16*)alloc(256 * 288 * 2), (_Float16*)alloc(256 * 256 * 2),
                      (_Float16*)alloc(256 * 256 * 2)};
  _Float16* wtV = (_Float16*)alloc(272 * 256 * 2);
  _Float16* wtH[2] = {(_Float16*)alloc(256 * 256 * 2), (_Float16*)alloc(128 * 256 * 2)};
  float* dists = (float*)alloc((size_t)B * N0 * 4);
  int* fpsidx = (int*)alloc((size_t)B * 2048 * 4);
  int* nidx = (int*)alloc((size_t)B * 2048 * 16 * 4);  // max entries across layers
  float* nx1 = (float*)alloc((size_t)B * 2048 * 3 * 4);
  float* nx2 = (float*)alloc((size_t)B * 1024 * 3 * 4);
  float* nx3 = (float*)alloc((size_t)B * 512 * 3 * 4);
  _Float16* feat1 = (_Float16*)alloc((size_t)B * 2048 * 128 * 2);
  _Float16* feat2 = (_Float16*)alloc((size_t)B * 1024 * 256 * 2);
  _Float16* feat3 = (_Float16*)alloc((size_t)B * 512 * 256 * 2);
  _Float16* pfeat = (_Float16*)alloc((size_t)B * 256 * 256 * 2);
  _Float16* arenaA = (_Float16*)alloc((size_t)21 << 20);  // X / even-stage outs
  _Float16* arenaB = (_Float16*)alloc((size_t)34 << 20);  // odd-stage outs

  auto conv = [&](const float* W, _Float16* Wt, int K, int N, int Kp, int Np) {
    int tot = Kp * Np;
    convert_wt_kernel<<<(tot + 255) / 256, 256, 0, stream>>>(W, Wt, K, N, Kp, Np);
  };
  auto gemm = [&](const _Float16* A, const _Float16* Wt, const float* bias, _Float16* Out,
                  int M, int Kpad, int Npad, int Nreal, bool relu, int ntiles) {
    if (ntiles == 4) {
      dim3 grid(M / 128, Npad / 64);
      if (relu)
        gemm_wmma_kernel<true, 4><<<grid, 256, 0, stream>>>(A, Wt, bias, Out, Kpad, Npad, Nreal);
      else
        gemm_wmma_kernel<false, 4><<<grid, 256, 0, stream>>>(A, Wt, bias, Out, Kpad, Npad, Nreal);
    } else {
      dim3 grid(M / 128, Npad / 16);
      if (relu)
        gemm_wmma_kernel<true, 1><<<grid, 256, 0, stream>>>(A, Wt, bias, Out, Kpad, Npad, Nreal);
      else
        gemm_wmma_kernel<false, 1><<<grid, 256, 0, stream>>>(A, Wt, bias, Out, Kpad, Npad, Nreal);
    }
  };

  // ---- weight conversion (f32 -> transposed padded f16) ----
  conv(sa1W[0], wt1[0], 3, 64, 32, 64);
  conv(sa1W[1], wt1[1], 64, 64, 64, 64);
  conv(sa1W[2], wt1[2], 64, 128, 64, 128);
  conv(sa2W[0], wt2[0], 131, 128, 160, 128);
  conv(sa2W[1], wt2[1], 128, 128, 128, 128);
  conv(sa2W[2], wt2[2], 128, 256, 128, 256);
  conv(sa3W[0], wt3[0], 259, 256, 288, 256);
  conv(sa3W[1], wt3[1], 256, 256, 256, 256);
  conv(sa3W[2], wt3[2], 256, 256, 256, 256);
  conv(voteW, wtV, 256, 259, 256, 272);
  conv(hW[0], wtH[0], 256, 256, 256, 256);
  conv(hW[1], wtH[1], 256, 128, 256, 128);

  // ---- SA1: 20000 -> 2048, nsample 16, MLP 3->64->64->128 ----
  fps_kernel<<<B, 1024, 0, stream>>>(pc, N0, 2048, fpsidx, dists);
  gather_xyz_kernel<<<(B * 2048 * 3 + 255) / 256, 256, 0, stream>>>(pc, fpsidx, nx1, 2048, N0,
                                                                    B * 2048 * 3);
  ball_query_kernel<<<(B * 2048 + 255) / 256, 256, 0, stream>>>(pc, nx1, N0, 2048, 16,
                                                                0.02f * 0.02f, nidx, B * 2048);
  {
    int tot = B * 2048 * 16 * 32;
    group_kernel<<<(tot + 255) / 256, 256, 0, stream>>>(pc, nx1, nidx, nullptr, 0, N0, 2048, 16,
                                                        32, arenaA, tot);
  }
  gemm(arenaA, wt1[0], sa1B[0], arenaB, 65536, 32, 64, 64, true, 4);
  gemm(arenaB, wt1[1], sa1B[1], arenaA, 65536, 64, 64, 64, true, 4);
  gemm(arenaA, wt1[2], sa1B[2], arenaB, 65536, 64, 128, 128, true, 4);
  maxpool_kernel<<<(B * 2048 * 128 + 255) / 256, 256, 0, stream>>>(arenaB, feat1, 2048, 16, 128,
                                                                   B * 2048 * 128);

  // ---- SA2: 2048 -> 1024, nsample 32, MLP 131->128->128->256 ----
  fps_kernel<<<B, 1024, 0, stream>>>(nx1, 2048, 1024, fpsidx, dists);
  gather_xyz_kernel<<<(B * 1024 * 3 + 255) / 256, 256, 0, stream>>>(nx1, fpsidx, nx2, 1024, 2048,
                                                                    B * 1024 * 3);
  ball_query_kernel<<<(B * 1024 + 255) / 256, 256, 0, stream>>>(nx1, nx2, 2048, 1024, 32,
                                                                0.04f * 0.04f, nidx, B * 1024);
  {
    int tot = B * 1024 * 32 * 160;
    group_kernel<<<(tot + 255) / 256, 256, 0, stream>>>(nx1, nx2, nidx, feat1, 128, 2048, 1024,
                                                        32, 160, arenaA, tot);
  }
  gemm(arenaA, wt2[0], sa2B[0], arenaB, 65536, 160, 128, 128, true, 4);
  gemm(arenaB, wt2[1], sa2B[1], arenaA, 65536, 128, 128, 128, true, 4);
  gemm(arenaA, wt2[2], sa2B[2], arenaB, 65536, 128, 256, 256, true, 4);
  maxpool_kernel<<<(B * 1024 * 256 + 255) / 256, 256, 0, stream>>>(arenaB, feat2, 1024, 32, 256,
                                                                   B * 1024 * 256);

  // ---- SA3: 1024 -> 512, nsample 32, MLP 259->256->256->256 ----
  fps_kernel<<<B, 1024, 0, stream>>>(nx2, 1024, 512, fpsidx, dists);
  gather_xyz_kernel<<<(B * 512 * 3 + 255) / 256, 256, 0, stream>>>(nx2, fpsidx, nx3, 512, 1024,
                                                                   B * 512 * 3);
  ball_query_kernel<<<(B * 512 + 255) / 256, 256, 0, stream>>>(nx2, nx3, 1024, 512, 32,
                                                               0.08f * 0.08f, nidx, B * 512);
  {
    int tot = B * 512 * 32 * 288;
    group_kernel<<<(tot + 255) / 256, 256, 0, stream>>>(nx2, nx3, nidx, feat2, 256, 1024, 512,
                                                        32, 288, arenaA, tot);
  }
  gemm(arenaA, wt3[0], sa3B[0], arenaB, 32768, 288, 256, 256, true, 4);
  gemm(arenaB, wt3[1], sa3B[1], arenaA, 32768, 256, 256, 256, true, 4);
  gemm(arenaA, wt3[2], sa3B[2], arenaB, 32768, 256, 256, 256, true, 4);
  maxpool_kernel<<<(B * 512 * 256 + 255) / 256, 256, 0, stream>>>(arenaB, feat3, 512, 32, 256,
                                                                  B * 512 * 256);

  // ---- vote (no relu; vote_xyz is dead code in the returned tuple) ----
  gemm(feat3, wtV, voteB, arenaA, 1024, 256, 272, 259, false, 1);
  permute_kernel<<<(B * 256 * 256 + 255) / 256, 256, 0, stream>>>(arenaA, perm, pfeat,
                                                                  B * 256 * 256);

  // ---- head MLP 256->256->128 ----
  gemm(pfeat, wtH[0], hB[0], arenaB, 512, 256, 256, 256, true, 4);
  gemm(arenaB, wtH[1], hB[1], arenaA, 512, 256, 128, 128, true, 4);

  // ---- output heads ----
  heads_kernel<<<(512 * 7 + 255) / 256, 256, 0, stream>>>(arenaA, objW, objB, cenW, cenB, sizW,
                                                          sizB, (float*)d_out, 512 * 7);
}
